// GMHA_10746008175403
// MI455X (gfx1250) — matrix operand order repaired and checked by reading, not compiled, not run on hardware
//
#include <hip/hip_runtime.h>
#include <hip/hip_bf16.h>

typedef __attribute__((ext_vector_type(16))) __bf16 bf16x16;
typedef __attribute__((ext_vector_type(8)))  __bf16 bf16x8;
typedef __attribute__((ext_vector_type(8)))  float  f32x8;

#define QLEN 1024
#define MLEN 1024
#define KLEN 2048
#define DMODEL 1024
#define NH 16
#define DH 64
#define SCALE_F 0.125f   // 1/sqrt(64)

// ---------------------------------------------------------------------------
// WMMA fragment loaders (CDNA5 16-bit layouts, CDNA5 ISA §7.12.2)
// A: 16x32 (MxK). lanes 0-15: M=lane, elems 0-7 -> K=0..7, elems 8-15 -> K=16..23
//                 lanes 16-31: same M, K offset +8 (and +24 for hi half).
// B: 32x16 (KxN) from row-major W[N][K]. lanes 0-15: N=lane, same K order as A;
//                 lanes 16-31: N=lane-16, K offset +8 (and +24 for hi half).
// ---------------------------------------------------------------------------
__device__ inline bf16x16 load_a_frag(const __bf16* __restrict__ A, int lda,
                                      int m0, int k0, int lane) {
    int m  = lane & 15;
    int kh = (lane >> 4) & 1;
    const __bf16* row = A + (long long)(m0 + m) * lda;
    bf16x8 lo = *(const bf16x8*)(row + k0 + 8 * kh);
    bf16x8 hi = *(const bf16x8*)(row + k0 + 16 + 8 * kh);
    bf16x16 a;
#pragma unroll
    for (int e = 0; e < 8; ++e) { a[e] = lo[e]; a[8 + e] = hi[e]; }
    return a;
}

__device__ inline bf16x16 load_b_frag(const __bf16* __restrict__ B, int ldb,
                                      int n0, int k0, int lane) {
    int n  = lane & 15;
    int kh = (lane >> 4) & 1;
    const __bf16* row = B + (long long)(n0 + n) * ldb;
    bf16x8 lo = *(const bf16x8*)(row + k0 + 8 * kh);
    bf16x8 hi = *(const bf16x8*)(row + k0 + 16 + 8 * kh);
    bf16x16 b;
#pragma unroll
    for (int e = 0; e < 8; ++e) { b[e] = lo[e]; b[8 + e] = hi[e]; }
    return b;
}

__device__ inline f32x8 wmma_bf16(bf16x16 a, bf16x16 b, f32x8 c) {
    return __builtin_amdgcn_wmma_f32_16x16x32_bf16(false, a, false, b,
                                                   (short)0, c, false, false);
}

// ---------------------------------------------------------------------------
// Generic bf16 GEMM: C[M,N] = A[M,K] * B[N,K]^T (+bias).
// 2x2 register-blocked: each wave owns a 32x32 output tile (4 accumulators),
// so 4 WMMAs per 4 fragment loads -> 2x the operand reuse of a 1x1 tile.
// 4 waves/block cover a 64x64 block tile. Batched via z-dim pointer strides;
// cstride lets us write the head-interleaved probs layout [i][j][h] directly.
// ---------------------------------------------------------------------------
__global__ void k_gemm(const __bf16* __restrict__ A, const __bf16* __restrict__ B,
                       float* __restrict__ C,
                       int M, int N, int K, int lda, int ldb, int ldc,
                       long long batchA, long long batchB, long long batchC,
                       int cstride, const float* __restrict__ bias)
{
    const int lane = threadIdx.x & 31;
    const int wave = threadIdx.x >> 5;
    const int m0 = blockIdx.y * 64 + (wave >> 1) * 32;
    const int n0 = blockIdx.x * 64 + (wave & 1) * 32;
    const int bz = blockIdx.z;
    if (n0 >= N || m0 >= M) return;

    const __bf16* Ab = A + (long long)bz * batchA;
    const __bf16* Bb = B + (long long)bz * batchB;

    f32x8 acc00 = {0.f, 0.f, 0.f, 0.f, 0.f, 0.f, 0.f, 0.f};
    f32x8 acc01 = acc00, acc10 = acc00, acc11 = acc00;

    for (int k0 = 0; k0 < K; k0 += 32) {
        bf16x16 a0 = load_a_frag(Ab, lda, m0,      k0, lane);
        bf16x16 a1 = load_a_frag(Ab, lda, m0 + 16, k0, lane);
        bf16x16 b0 = load_b_frag(Bb, ldb, n0,      k0, lane);
        bf16x16 b1 = load_b_frag(Bb, ldb, n0 + 16, k0, lane);
        acc00 = wmma_bf16(a0, b0, acc00);
        acc01 = wmma_bf16(a0, b1, acc01);
        acc10 = wmma_bf16(a1, b0, acc10);
        acc11 = wmma_bf16(a1, b1, acc11);
    }

    const int n  = lane & 15;
    const int mh = lane >> 4;
    float* Cb = C + (long long)bz * batchC;
    const float bv0 = bias ? bias[n0 + n]      : 0.f;
    const float bv1 = bias ? bias[n0 + 16 + n] : 0.f;
#pragma unroll
    for (int r = 0; r < 8; ++r) {
        long long ma = m0 + r + 8 * mh;
        long long mb = ma + 16;
        Cb[(ma * ldc + (n0 + n))      * (long long)cstride] = acc00[r] + bv0;
        Cb[(ma * ldc + (n0 + 16 + n)) * (long long)cstride] = acc01[r] + bv1;
        Cb[(mb * ldc + (n0 + n))      * (long long)cstride] = acc10[r] + bv0;
        Cb[(mb * ldc + (n0 + 16 + n)) * (long long)cstride] = acc11[r] + bv1;
    }
}

// ---------------------------------------------------------------------------
// BD term: T = (Q+v)_h x RK_h^T (WMMA), scatter-added at the Transformer-XL
// shifted column: scores[i][jr-1023+i][h] += T[i][jr].
// The shift's wrap region coincides exactly with the causal mask, so simple
// bounds checks are exact.
// ---------------------------------------------------------------------------
__global__ void k_bd_add(const __bf16* __restrict__ QV, const __bf16* __restrict__ RK,
                         float* __restrict__ S)
{
    const int lane = threadIdx.x & 31;
    const int wave = threadIdx.x >> 5;
    const int jr0 = (blockIdx.x * 4 + wave) * 16;
    const int m0  = blockIdx.y * 16;
    const int h   = blockIdx.z;

    const __bf16* A = QV + (long long)h * QLEN * DH;
    const __bf16* B = RK + (long long)h * KLEN * DH;

    f32x8 acc = {0.f, 0.f, 0.f, 0.f, 0.f, 0.f, 0.f, 0.f};
    for (int k0 = 0; k0 < DH; k0 += 32) {
        bf16x16 a = load_a_frag(A, DH, m0, k0, lane);
        bf16x16 b = load_b_frag(B, DH, jr0, k0, lane);
        acc = wmma_bf16(a, b, acc);
    }

    const int n  = lane & 15;
    const int mh = lane >> 4;
#pragma unroll
    for (int r = 0; r < 8; ++r) {
        int i  = m0 + r + 8 * mh;
        int jr = jr0 + n;
        int j  = jr - (QLEN - 1) + i;      // rel_shift target column
        if (j >= 0 && j < KLEN) {
            long long idx = ((long long)i * KLEN + j) * NH + h;
            S[idx] += acc[r];
        }
    }
}

// ---------------------------------------------------------------------------
// Masked softmax over klen per (i,h); one wave per row. Writes normalized
// probs back in-place (f32, the reference output) and a bf16 head-major copy
// P[h][i][j] for the WMMA attn_vec GEMM.
// ---------------------------------------------------------------------------
__global__ void k_softmax(float* __restrict__ S, __bf16* __restrict__ P)
{
    const int lane = threadIdx.x & 31;
    const int wave = threadIdx.x >> 5;
    const int row  = blockIdx.x * 8 + wave;        // 0 .. QLEN*NH-1
    if (row >= QLEN * NH) return;
    const int i = row >> 4;
    const int h = row & 15;
    const int jmax = i + MLEN;                     // valid: j <= i + MLEN

    float mx = -3.0e38f;
    for (int j = lane; j < KLEN; j += 32) {
        float s = S[((long long)i * KLEN + j) * NH + h] * SCALE_F;
        if (j <= jmax) mx = fmaxf(mx, s);
    }
#pragma unroll
    for (int off = 16; off; off >>= 1) mx = fmaxf(mx, __shfl_xor(mx, off, 32));

    float sum = 0.f;
    for (int j = lane; j < KLEN; j += 32) {
        float s = S[((long long)i * KLEN + j) * NH + h] * SCALE_F;
        sum += (j <= jmax) ? __expf(s - mx) : 0.f;
    }
#pragma unroll
    for (int off = 16; off; off >>= 1) sum += __shfl_xor(sum, off, 32);
    const float inv = 1.f / sum;

    for (int j = lane; j < KLEN; j += 32) {
        long long idx = ((long long)i * KLEN + j) * NH + h;
        float s = S[idx] * SCALE_F;
        float p = (j <= jmax) ? __expf(s - mx) * inv : 0.f;
        S[idx] = p;
        P[((long long)h * QLEN + i) * KLEN + j] = (__bf16)p;
    }
}

// f32 -> bf16 conversion
__global__ void k_cvt(const float* __restrict__ src, __bf16* __restrict__ dst,
                      long long n)
{
    long long t = (long long)blockIdx.x * blockDim.x + threadIdx.x;
    if (t < n) dst[t] = (__bf16)src[t];
}

// ---------------------------------------------------------------------------
// Split w_heads into head-major bf16 operands: QU/QV [h][i][d] (q + u/v),
// K [h][j][d], V^T [h][d][j], and head-major RK [h][j][d] from r @ W_r^T.
// ---------------------------------------------------------------------------
__global__ void k_prep(const float* __restrict__ wh, const float* __restrict__ rk,
                       const float* __restrict__ u, const float* __restrict__ v,
                       __bf16* __restrict__ qu, __bf16* __restrict__ qv,
                       __bf16* __restrict__ kh, __bf16* __restrict__ vT,
                       __bf16* __restrict__ rkh)
{
    long long tid = (long long)blockIdx.x * blockDim.x + threadIdx.x;
    if (tid >= (long long)KLEN * DMODEL) return;
    const int col = (int)(tid & (DMODEL - 1));   // h*64 + d
    const int j   = (int)(tid >> 10);            // 0..KLEN-1
    const int h   = col >> 6;
    const int d   = col & 63;
    const long long whrow = (long long)j * (3 * DMODEL);

    kh [((long long)h * KLEN + j) * DH + d] = (__bf16)wh[whrow + DMODEL + col];
    vT [((long long)h * DH + d) * KLEN + j] = (__bf16)wh[whrow + 2 * DMODEL + col];
    rkh[((long long)h * KLEN + j) * DH + d] = (__bf16)rk[(long long)j * DMODEL + col];
    if (j >= MLEN) {
        const int i = j - MLEN;
        float q = wh[whrow + col];
        qu[((long long)h * QLEN + i) * DH + d] = (__bf16)(q + u[col]);
        qv[((long long)h * QLEN + i) * DH + d] = (__bf16)(q + v[col]);
    }
}

// ---------------------------------------------------------------------------
// GRU: inherently sequential over time -> single 1024-thread workgroup
// (32 waves on one WGP), hidden state in LDS, bf16 W_hh streamed from L2.
// PyTorch gate order r,z,n. Thread c owns hidden column c and gate rows
// {c, 1024+c, 2048+c}.
// ---------------------------------------------------------------------------
__global__ void __launch_bounds__(1024)
k_gru(const float* __restrict__ xproj, const __bf16* __restrict__ Whh,
      const float* __restrict__ bhh, const float* __restrict__ h0,
      float* __restrict__ ys)
{
    __shared__ float hs[DMODEL];
    const int c = threadIdx.x;
    hs[c] = h0[c];
    __syncthreads();

    const __bf16* wr = Whh + (long long)c * DMODEL;
    const __bf16* wz = Whh + (long long)(DMODEL + c) * DMODEL;
    const __bf16* wn = Whh + (long long)(2 * DMODEL + c) * DMODEL;
    const float br = bhh[c], bz = bhh[DMODEL + c], bn = bhh[2 * DMODEL + c];

    for (int t = 0; t < QLEN; ++t) {
        float gr = br, gz = bz, gn = bn;
        for (int d = 0; d < DMODEL; ++d) {
            float hd = hs[d];
            gr = fmaf(hd, (float)wr[d], gr);
            gz = fmaf(hd, (float)wz[d], gz);
            gn = fmaf(hd, (float)wn[d], gn);
        }
        const float* xt = xproj + (long long)t * (3 * DMODEL);
        float rg = 1.f / (1.f + __expf(-(xt[c] + gr)));
        float zg = 1.f / (1.f + __expf(-(xt[DMODEL + c] + gz)));
        float ng = tanhf(xt[2 * DMODEL + c] + rg * gn);
        float hn = (1.f - zg) * ng + zg * hs[c];
        __syncthreads();
        hs[c] = hn;
        ys[(long long)t * DMODEL + c] = hn;
        __syncthreads();
    }
}

// ---------------------------------------------------------------------------
extern "C" void kernel_launch(void* const* d_in, const int* in_sizes, int n_in,
                              void* d_out, int out_size, void* d_ws, size_t ws_size,
                              hipStream_t stream)
{
    const float* inputs = (const float*)d_in[0];
    const float* r      = (const float*)d_in[1];
    const float* u      = (const float*)d_in[2];
    const float* v      = (const float*)d_in[3];
    /* d_in[4] = attn_mask: mask is computed analytically (causal, k=MLEN+1) */
    const float* mem    = (const float*)d_in[5];
    const float* W_qkv  = (const float*)d_in[6];
    const float* W_r    = (const float*)d_in[7];
    const float* W_o    = (const float*)d_in[8];
    const float* W_ih   = (const float*)d_in[9];
    const float* W_hh   = (const float*)d_in[10];
    const float* b_ih   = (const float*)d_in[11];
    const float* b_hh   = (const float*)d_in[12];
    const float* h0     = (const float*)d_in[13];

    float* ys     = (float*)d_out;                              // [1024,1,1024]
    float* scores = (float*)d_out + (long long)QLEN * DMODEL;   // probs [q,k,1,n]

    char* w = (char*)d_ws;
    auto alloc = [&](size_t bytes) -> char* {
        char* p = w;
        w += (bytes + 255) & ~(size_t)255;
        return p;
    };

    __bf16* cat_b  = (__bf16*)alloc((size_t)KLEN * DMODEL * 2);
    __bf16* r_b    = (__bf16*)alloc((size_t)KLEN * DMODEL * 2);
    __bf16* wqkv_b = (__bf16*)alloc((size_t)3 * DMODEL * DMODEL * 2);
    __bf16* wr_b   = (__bf16*)alloc((size_t)DMODEL * DMODEL * 2);
    __bf16* wo_b   = (__bf16*)alloc((size_t)DMODEL * DMODEL * 2);
    __bf16* wih_b  = (__bf16*)alloc((size_t)3 * DMODEL * DMODEL * 2);
    __bf16* whh_b  = (__bf16*)alloc((size_t)3 * DMODEL * DMODEL * 2);
    float*  wheads = (float*) alloc((size_t)KLEN * 3 * DMODEL * 4);
    float*  rk_f   = (float*) alloc((size_t)KLEN * DMODEL * 4);
    __bf16* qu_b   = (__bf16*)alloc((size_t)NH * QLEN * DH * 2);
    __bf16* qv_b   = (__bf16*)alloc((size_t)NH * QLEN * DH * 2);
    __bf16* k_b    = (__bf16*)alloc((size_t)NH * KLEN * DH * 2);
    __bf16* vT_b   = (__bf16*)alloc((size_t)NH * DH * KLEN * 2);
    __bf16* rk_b   = (__bf16*)alloc((size_t)NH * KLEN * DH * 2);
    __bf16* p_b    = (__bf16*)alloc((size_t)NH * QLEN * KLEN * 2);
    float*  av_f   = (float*) alloc((size_t)QLEN * DMODEL * 4);
    __bf16* av_b   = (__bf16*)alloc((size_t)QLEN * DMODEL * 2);
    float*  y_f    = (float*) alloc((size_t)QLEN * DMODEL * 4);
    __bf16* y_b    = (__bf16*)alloc((size_t)QLEN * DMODEL * 2);
    float*  xp_f   = (float*) alloc((size_t)QLEN * 3 * DMODEL * 4);

    auto cvt = [&](const float* s, __bf16* d, long long n) {
        k_cvt<<<dim3((unsigned)((n + 255) / 256)), dim3(256), 0, stream>>>(s, d, n);
    };
    auto gemm = [&](const __bf16* A, const __bf16* B, float* C, int M, int N, int K,
                    int lda, int ldb, int ldc, long long bA, long long bB,
                    long long bC, int cstride, int nb, const float* bias) {
        k_gemm<<<dim3(N / 64, M / 64, nb), dim3(128), 0, stream>>>(
            A, B, C, M, N, K, lda, ldb, ldc, bA, bB, bC, cstride, bias);
    };

    // 1) bf16 conversions (cat = [mem ; inputs])
    cvt(mem,    cat_b,                            (long long)MLEN * DMODEL);
    cvt(inputs, cat_b + (long long)MLEN * DMODEL, (long long)QLEN * DMODEL);
    cvt(r,      r_b,    (long long)KLEN * DMODEL);
    cvt(W_qkv,  wqkv_b, (long long)3 * DMODEL * DMODEL);
    cvt(W_r,    wr_b,   (long long)DMODEL * DMODEL);
    cvt(W_o,    wo_b,   (long long)DMODEL * DMODEL);
    cvt(W_ih,   wih_b,  (long long)3 * DMODEL * DMODEL);
    cvt(W_hh,   whh_b,  (long long)3 * DMODEL * DMODEL);

    // 2) w_heads = cat @ W_qkv^T   [2048, 3072]
    gemm(cat_b, wqkv_b, wheads, KLEN, 3 * DMODEL, DMODEL,
         DMODEL, DMODEL, 3 * DMODEL, 0, 0, 0, 1, 1, nullptr);
    // 3) r_k = r @ W_r^T           [2048, 1024]
    gemm(r_b, wr_b, rk_f, KLEN, DMODEL, DMODEL,
         DMODEL, DMODEL, DMODEL, 0, 0, 0, 1, 1, nullptr);

    // 4) head-major bf16 operands
    k_prep<<<dim3((KLEN * DMODEL) / 256), dim3(256), 0, stream>>>(
        wheads, rk_f, u, v, qu_b, qv_b, k_b, vT_b, rk_b);

    // 5) AC content scores -> probs buffer, layout [i][j][h]
    gemm(qu_b, k_b, scores, QLEN, KLEN, DH,
         DH, DH, KLEN, (long long)QLEN * DH, (long long)KLEN * DH, 1, NH, NH, nullptr);

    // 6) BD position scores, rel-shifted scatter-add
    k_bd_add<<<dim3(KLEN / 64, QLEN / 16, NH), dim3(128), 0, stream>>>(
        qv_b, rk_b, scores);

    // 7) masked softmax (writes probs f32 in-place + bf16 head-major copy)
    k_softmax<<<dim3((QLEN * NH) / 8), dim3(256), 0, stream>>>(scores, p_b);

    // 8) attn_vec = P @ V  -> av[i][h*64+d]
    gemm(p_b, vT_b, av_f, QLEN, DH, KLEN,
         KLEN, KLEN, DMODEL, (long long)QLEN * KLEN, (long long)DH * KLEN,
         DH, 1, NH, nullptr);

    // 9) y = attn_vec @ W_o^T
    cvt(av_f, av_b, (long long)QLEN * DMODEL);
    gemm(av_b, wo_b, y_f, QLEN, DMODEL, DMODEL,
         DMODEL, DMODEL, DMODEL, 0, 0, 0, 1, 1, nullptr);

    // 10) x_proj = y @ W_ih^T + b_ih
    cvt(y_f, y_b, (long long)QLEN * DMODEL);
    gemm(y_b, wih_b, xp_f, QLEN, 3 * DMODEL, DMODEL,
         DMODEL, DMODEL, 3 * DMODEL, 0, 0, 0, 1, 1, b_ih);

    // 11) GRU scan -> ys
    k_gru<<<dim3(1), dim3(1024), 0, stream>>>(xp_f, whh_b, b_hh, h0, ys);

    (void)in_sizes; (void)n_in; (void)out_size; (void)ws_size;
}
